// InterSliceSelfAttention_5952824672942
// MI455X (gfx1250) — matrix operand order repaired and checked by reading, not compiled, not run on hardware
//
#include <hip/hip_runtime.h>

typedef __attribute__((ext_vector_type(16))) __bf16       v16bf;
typedef __attribute__((ext_vector_type(2)))  __bf16       v2bf;
typedef __attribute__((ext_vector_type(8)))  float        v8f;
typedef __attribute__((ext_vector_type(8)))  unsigned int v8u;

#define WMMA_BF16(A, Bm, Cm)                                                   \
  __builtin_amdgcn_wmma_f32_16x16x32_bf16(false, __builtin_bit_cast(v16bf,(A)),\
                                          false, __builtin_bit_cast(v16bf,(Bm)),\
                                          (short)0, (Cm), false, false)

// Problem shape (fixed by the reference)
constexpr int Bsz = 2, Cc = 128, Dd = 32, Hh = 96, Ww = 96;
constexpr int WT = 16, NW = 16, THREADS = NW * 32;
constexpr int sH = Ww;              // 96
constexpr int sD = Hh * Ww;         // 9216
constexpr int sC = Dd * sD;         // 294912
constexpr int sB = Cc * sC;         // 37748736

// Per-wave LDS slab (ushort units). Padded pitches -> conflict-free 16-lane
// fragment accesses. All post-projection staging reuses the dead P region.
constexpr int P_PITCH  = 136;       // P: [d=32][c=128+pad], [0, 4352)
constexpr int QK_PITCH = 68;        // K stage: [e=32][q=64+pad], [0, 2176) after P dies
constexpr int AT_PITCH = 36;        // attn: [d=32][e=32+pad]
constexpr int OFF_ATTN = 2176;      // [2176, 3328)
constexpr int OFF_OUT  = 3328;      // f32 out staging, 512 f32 = [3328, 4352)
constexpr int OFF_V    = 4352;      // V-tile transpose scratch [4352, 4928)
constexpr int SLAB     = 4930;      // 9860 B; word stride 2465 (odd) across slabs

// Hardware bf16 converts (RNE): let clang lower fptrunc -> v_cvt_*bf16_f32.
__device__ __forceinline__ unsigned short bf16_1(float a) {
  return __builtin_bit_cast(unsigned short, (__bf16)a);
}
__device__ __forceinline__ unsigned pk_bf16(float a, float b) {
  v2bf p; p[0] = (__bf16)a; p[1] = (__bf16)b;
  return __builtin_bit_cast(unsigned, p);
}

// A-fragment (16xK32 bf16) from f32 row-major weights in global memory.
// Lane l holds row m=l&15; vgpr j -> K = (j<4?0:16)+8*hi+2*(j&3) (pairs).
__device__ __forceinline__ v8u afrag_w(const float* __restrict__ wrow, int kc, int hi) {
  v8u f;
#pragma unroll
  for (int j = 0; j < 8; ++j) {
    int k = kc + ((j < 4) ? 0 : 16) + 8 * hi + 2 * (j & 3);
    f[j] = pk_bf16(wrow[k], wrow[k + 1]);
  }
  return f;
}
// A-fragment from LDS (bf16 pairs contiguous along K at fixed row).
__device__ __forceinline__ v8u afrag_lds(const unsigned short* base, int pitch,
                                         int row, int kb, int hi) {
  v8u f; const unsigned short* r = base + row * pitch + kb + 8 * hi;
#pragma unroll
  for (int j = 0; j < 8; ++j) {
    int k = ((j < 4) ? 0 : 16) + 2 * (j & 3);
    f[j] = *reinterpret_cast<const unsigned int*>(r + k);
  }
  return f;
}
// B-fragment (K32x16 bf16) from LDS: lane l holds col n=l&15; vgpr j -> K = kb+(j<4?0:16)+2*(j&3).
__device__ __forceinline__ v8u bfrag_lds(const unsigned short* base, int pitch,
                                         int row, int kb) {
  v8u f; const unsigned short* r = base + row * pitch + kb;
#pragma unroll
  for (int j = 0; j < 8; ++j)
    f[j] = *reinterpret_cast<const unsigned int*>(r + ((j < 4) ? 0 : 16) + 2 * (j & 3));
  return f;
}

__global__ __launch_bounds__(THREADS, 1)
void isa_fused(const float* __restrict__ x,  const float* __restrict__ P,
               const float* __restrict__ Wq, const float* __restrict__ bq,
               const float* __restrict__ Wk, const float* __restrict__ bk,
               const float* __restrict__ Wv, const float* __restrict__ bv,
               const float* __restrict__ pe, const float* __restrict__ gmp,
               float* __restrict__ out) {
  __shared__ unsigned short smem[NW * SLAB];

  const int t    = threadIdx.x;
  const int lane = t & 31, wave = t >> 5;
  const int lo   = lane & 15, hi = lane >> 4;
  const int bi   = blockIdx.x;
  const int wt   = bi % (Ww / WT);
  const int h    = (bi / (Ww / WT)) % Hh;
  const int b    = bi / ((Ww / WT) * Hh);
  const int gbase = b * sB + h * sH + wt * WT;   // + c*sC + d*sD + w

  // ---- Phase A: cooperative, coalesced load of P (128c x 32d x 16w) -> LDS bf16 [w][d][c]
  {
    const int w = t & 15;                       // 16 consecutive w => 64B runs
    unsigned short* slw = smem + w * SLAB;
    const int s0 = t >> 4;                      // 32 slot groups
    for (int i = 0; i < 128; ++i) {
      int s = s0 + 32 * i;                      // 4096 (c,d) slots
      int c = s >> 5, d = s & 31;
      float v = __builtin_nontemporal_load(&P[gbase + c * sC + d * sD + w]);
      slw[d * P_PITCH + c] = bf16_1(v);
    }
  }
  __syncthreads();

  unsigned short* slab = smem + wave * SLAB;    // this wave's private slab

  // ---- Phase B1: V = Wv(128x128) * P(128x32); keep V as bf16 A-frags (o x e)
  v8u vfrag[8];
  {
    v8f vacc[8][2];
#pragma unroll
    for (int ot = 0; ot < 8; ++ot)
#pragma unroll
      for (int et = 0; et < 2; ++et) vacc[ot][et] = v8f{};
#pragma unroll
    for (int ks = 0; ks < 4; ++ks) {
      v8u bf[2];
#pragma unroll
      for (int et = 0; et < 2; ++et)
        bf[et] = bfrag_lds(slab, P_PITCH, et * 16 + lo, ks * 32 + 8 * hi);
#pragma unroll
      for (int ot = 0; ot < 8; ++ot) {
        v8u af = afrag_w(Wv + (ot * 16 + lo) * Cc, ks * 32, hi);
#pragma unroll
        for (int et = 0; et < 2; ++et)
          vacc[ot][et] = WMMA_BF16(af, bf[et], vacc[ot][et]);
      }
    }
#pragma unroll
    for (int ot = 0; ot < 8; ++ot) {            // C/D -> A transpose via LDS scratch
#pragma unroll
      for (int et = 0; et < 2; ++et)
#pragma unroll
        for (int j = 0; j < 8; ++j) {
          int o = ot * 16 + j + 8 * hi;
          slab[OFF_V + (j + 8 * hi) * AT_PITCH + et * 16 + lo] =
              bf16_1(vacc[ot][et][j] + bv[o]);
        }
      vfrag[ot] = afrag_lds(slab + OFF_V, AT_PITCH, lo, 0, hi);
    }
  }

  // ---- Phase B2q: Q projection; fold straight into A-frags of Q^T IN REGISTERS.
  // A-frag of Q^T: lane(lo,hi), vgpr j holds Q[q0,d],Q[q0+1,d] with d=dt*16+lo,
  // q0 = qs*32 + (j<4?0:16) + 8*hi + 2*(j&3).  Those rows are exactly this
  // lane's own C/D rows (vgpr 2*(j&3) of tile qt=2qs+(j>=4)) -> pure in-lane perm.
  v8u aq[2][2];                                  // [dt][qs]
  {
    v8f qacc[4][2];
#pragma unroll
    for (int qt = 0; qt < 4; ++qt)
#pragma unroll
      for (int dt = 0; dt < 2; ++dt) qacc[qt][dt] = v8f{};
#pragma unroll
    for (int ks = 0; ks < 4; ++ks) {
      v8u bf[2];
#pragma unroll
      for (int dt = 0; dt < 2; ++dt)
        bf[dt] = bfrag_lds(slab, P_PITCH, dt * 16 + lo, ks * 32 + 8 * hi);
#pragma unroll
      for (int qt = 0; qt < 4; ++qt) {
        v8u af = afrag_w(Wq + (qt * 16 + lo) * Cc, ks * 32, hi);
#pragma unroll
        for (int dt = 0; dt < 2; ++dt)
          qacc[qt][dt] = WMMA_BF16(af, bf[dt], qacc[qt][dt]);
      }
    }
#pragma unroll
    for (int dt = 0; dt < 2; ++dt) {
      int d = dt * 16 + lo;
#pragma unroll
      for (int qs = 0; qs < 2; ++qs) {
        v8u f;
#pragma unroll
        for (int j = 0; j < 8; ++j) {
          int qt = 2 * qs + ((j >= 4) ? 1 : 0);
          int js = 2 * (j & 3);
          int q0 = qt * 16 + 8 * hi + js;
          f[j] = pk_bf16(qacc[qt][dt][js]     + bq[q0]     + pe[q0 * Dd + d],
                         qacc[qt][dt][js + 1] + bq[q0 + 1] + pe[(q0 + 1) * Dd + d]);
        }
        aq[dt][qs] = f;
      }
    }
  }

  // ---- Phase B2k: K projection (last P reader); stage bf16 [e][q] over dead P.
  {
    v8f kacc[4][2];
#pragma unroll
    for (int qt = 0; qt < 4; ++qt)
#pragma unroll
      for (int et = 0; et < 2; ++et) kacc[qt][et] = v8f{};
#pragma unroll
    for (int ks = 0; ks < 4; ++ks) {
      v8u bf[2];
#pragma unroll
      for (int et = 0; et < 2; ++et)
        bf[et] = bfrag_lds(slab, P_PITCH, et * 16 + lo, ks * 32 + 8 * hi);
#pragma unroll
      for (int qt = 0; qt < 4; ++qt) {
        v8u af = afrag_w(Wk + (qt * 16 + lo) * Cc, ks * 32, hi);
#pragma unroll
        for (int et = 0; et < 2; ++et)
          kacc[qt][et] = WMMA_BF16(af, bf[et], kacc[qt][et]);
      }
    }
#pragma unroll
    for (int qt = 0; qt < 4; ++qt)
#pragma unroll
      for (int et = 0; et < 2; ++et) {
        int e = et * 16 + lo;
        unsigned short* krow = slab + e * QK_PITCH + qt * 16 + 8 * hi;
#pragma unroll
        for (int jj = 0; jj < 4; ++jj) {
          int q0 = qt * 16 + 8 * hi + 2 * jj;   // C/D row = j + 8*hi
          *reinterpret_cast<unsigned int*>(krow + 2 * jj) =
              pk_bf16(kacc[qt][et][2 * jj]     + bk[q0]     + pe[q0 * Dd + e],
                      kacc[qt][et][2 * jj + 1] + bk[q0 + 1] + pe[(q0 + 1) * Dd + e]);
        }
      }
  }

  // ---- Phase B3: attn = Q^T K (32x32, contract q=64), scale, softmax over e
  v8u battn[2];
  {
    v8f aacc[2][2];
#pragma unroll
    for (int dt = 0; dt < 2; ++dt)
#pragma unroll
      for (int et = 0; et < 2; ++et) aacc[dt][et] = v8f{};
#pragma unroll
    for (int qs = 0; qs < 2; ++qs) {
      v8u bk2[2];
#pragma unroll
      for (int et = 0; et < 2; ++et)
        bk2[et] = bfrag_lds(slab, QK_PITCH, et * 16 + lo, qs * 32 + 8 * hi);
#pragma unroll
      for (int dt = 0; dt < 2; ++dt)
#pragma unroll
        for (int et = 0; et < 2; ++et)
          aacc[dt][et] = WMMA_BF16(aq[dt][qs], bk2[et], aacc[dt][et]);
    }
    // rows (d) live in 16-lane halves of the C/D layout -> width-16 reductions
#pragma unroll
    for (int dt = 0; dt < 2; ++dt)
#pragma unroll
      for (int j = 0; j < 8; ++j) {
        float v0 = aacc[dt][0][j] * 0.125f, v1 = aacc[dt][1][j] * 0.125f;
        float m = fmaxf(v0, v1);
#pragma unroll
        for (int s = 8; s >= 1; s >>= 1) m = fmaxf(m, __shfl_xor(m, s, 16));
        float e0 = __expf(v0 - m), e1 = __expf(v1 - m);
        float sum = e0 + e1;
#pragma unroll
        for (int s = 8; s >= 1; s >>= 1) sum += __shfl_xor(sum, s, 16);
        float r = 1.0f / sum;
        aacc[dt][0][j] = e0 * r; aacc[dt][1][j] = e1 * r;
      }
    // stage attn bf16 as [d][e]; reload as B-frags of attn^T
#pragma unroll
    for (int dt = 0; dt < 2; ++dt)
#pragma unroll
      for (int et = 0; et < 2; ++et)
#pragma unroll
        for (int j = 0; j < 8; ++j)
          slab[OFF_ATTN + (dt * 16 + j + 8 * hi) * AT_PITCH + et * 16 + lo] =
              bf16_1(aacc[dt][et][j]);
#pragma unroll
    for (int dt = 0; dt < 2; ++dt)
      battn[dt] = bfrag_lds(slab + OFF_ATTN, AT_PITCH, dt * 16 + lo, 8 * hi);
  }

  // ---- Phase B4: out = V * attn^T, staged epilogue with coalesced blended stores
  const float g  = 1.0f / (1.0f + __expf(-gmp[0]));
  const float og = 1.0f - g;
  float* stg_my = reinterpret_cast<float*>(slab + OFF_OUT);
#pragma unroll
  for (int ot = 0; ot < 8; ++ot) {
    v8f oacc[2];
#pragma unroll
    for (int dt = 0; dt < 2; ++dt) {
      v8f z{};
      oacc[dt] = WMMA_BF16(vfrag[ot], battn[dt], z);
    }
#pragma unroll
    for (int dt = 0; dt < 2; ++dt)
#pragma unroll
      for (int j = 0; j < 8; ++j)
        stg_my[(j + 8 * hi) * 32 + dt * 16 + lo] = oacc[dt][j];
    __syncthreads();
    {
      const int w = t & 15;                     // coalesced: 16 consecutive w
      const int s0 = t >> 4;
#pragma unroll
      for (int i = 0; i < 16; ++i) {
        int slot = s0 + 32 * i;                 // 512 (o,d) slots
        int ol = slot >> 5, d = slot & 31;
        const float* stg = reinterpret_cast<const float*>(smem + w * SLAB + OFF_OUT);
        float v = stg[ol * 32 + d];
        int gi = gbase + (ot * 16 + ol) * sC + d * sD + w;
        float xr = __builtin_nontemporal_load(&x[gi]);
        __builtin_nontemporal_store(g * v + og * xr, &out[gi]);
      }
    }
    __syncthreads();
  }
}

extern "C" void kernel_launch(void* const* d_in, const int* in_sizes, int n_in,
                              void* d_out, int out_size, void* d_ws, size_t ws_size,
                              hipStream_t stream) {
  (void)in_sizes; (void)n_in; (void)out_size; (void)d_ws; (void)ws_size;
  const float* x  = (const float*)d_in[0];
  const float* p  = (const float*)d_in[1];
  const float* Wq = (const float*)d_in[2];
  const float* bq = (const float*)d_in[3];
  const float* Wk = (const float*)d_in[4];
  const float* bk = (const float*)d_in[5];
  const float* Wv = (const float*)d_in[6];
  const float* bv = (const float*)d_in[7];
  const float* pe = (const float*)d_in[8];
  const float* gm = (const float*)d_in[9];
  float* out = (float*)d_out;
  dim3 grid(Bsz * Hh * (Ww / WT));               // 1152 blocks
  isa_fused<<<grid, THREADS, 0, stream>>>(x, p, Wq, bq, Wk, bk, Wv, bv, pe, gm, out);
}
